// Attention_12764642804013
// MI455X (gfx1250) — compile-verified, matrix operand order checked by reading, not run on hardware
//
#include <hip/hip_runtime.h>
#include <hip/hip_bf16.h>
#include <math.h>

// ---------------------------------------------------------------------------
// Types / WMMA helpers (CDNA5 gfx1250, wave32)
// ---------------------------------------------------------------------------
typedef __attribute__((ext_vector_type(16))) __bf16 v16bf;
typedef __attribute__((ext_vector_type(8)))  float  v8f;

#define DEV static __device__ __forceinline__

DEV v8f vzero8() {
  v8f z;
#pragma unroll
  for (int i = 0; i < 8; ++i) z[i] = 0.f;
  return z;
}

DEV v8f wmma_bf16(v16bf a, v16bf b, v8f c) {
  return __builtin_amdgcn_wmma_f32_16x16x32_bf16(
      /*neg_a=*/false, a, /*neg_b=*/false, b,
      /*c_mod=*/(short)0, c, /*reuse_a=*/false, /*reuse_b=*/false);
}

// A-style 16x32 bf16 fragment from a row-major (16 x K) tile.
// lane<16: M=lane,    K in {kb..kb+7, kb+16..kb+23} with kb=0
// lane>=16:M=lane-16, same with kb=8
DEV v16bf frag_A_rowmajor(const __bf16* __restrict__ p, int ld, int lane) {
  const int row = lane & 15;
  const int kb  = (lane & 16) ? 8 : 0;
  const __bf16* r = p + (size_t)row * ld + kb;
  v16bf f;
#pragma unroll
  for (int e = 0; e < 8; ++e) f[e] = r[e];
#pragma unroll
  for (int e = 0; e < 8; ++e) f[8 + e] = r[16 + e];
  return f;
}

// B operand (K=32 x N=16 tile) gathered from a row-major K x N storage.
// lane holds column n = lane&15, K elements {kb..kb+7, kb+16..kb+23}.
DEV v16bf frag_B_kxn(const __bf16* __restrict__ p, int ld, int lane) {
  const int n  = lane & 15;
  const int kb = (lane & 16) ? 8 : 0;
  v16bf f;
#pragma unroll
  for (int e = 0; e < 8; ++e) f[e] = p[(size_t)(kb + e) * ld + n];
#pragma unroll
  for (int e = 0; e < 8; ++e) f[8 + e] = p[(size_t)(kb + 16 + e) * ld + n];
  return f;
}

// ---------------------------------------------------------------------------
// Generic tiled GEMM: C(Mx128) = A(MxK) * W(128xK)^T (+bias), WMMA bf16.
// One workgroup: 64 rows x 128 cols, 4 waves, K staged in LDS chunks of 32.
// ---------------------------------------------------------------------------
template <typename TA, typename TB>
__global__ __launch_bounds__(128)
void gemm_n128(const TA* __restrict__ A, const TB* __restrict__ W,
               const float* __restrict__ bias, float* __restrict__ C, int K) {
  __shared__ __bf16 As[64 * 32];
  __shared__ __bf16 Bs[128 * 32];
  const int tid  = threadIdx.x;
  const int lane = tid & 31;
  const int wv   = tid >> 5;
  const size_t row0 = (size_t)blockIdx.x * 64;

  v8f acc[8];
#pragma unroll
  for (int j = 0; j < 8; ++j) acc[j] = vzero8();

  for (int k0 = 0; k0 < K; k0 += 32) {
    for (int i = tid; i < 64 * 32; i += 128) {
      int r = i >> 5, c = i & 31;
      As[i] = (__bf16)(float)A[(row0 + r) * (size_t)K + k0 + c];
    }
    for (int i = tid; i < 128 * 32; i += 128) {
      int r = i >> 5, c = i & 31;
      Bs[i] = (__bf16)(float)W[(size_t)r * K + k0 + c];
    }
    __syncthreads();
    v16bf a = frag_A_rowmajor(&As[wv * 16 * 32], 32, lane);
#pragma unroll
    for (int j = 0; j < 8; ++j) {
      v16bf b = frag_A_rowmajor(&Bs[j * 16 * 32], 32, lane);  // W rows = B^T
      acc[j] = wmma_bf16(a, b, acc[j]);
    }
    __syncthreads();
  }
  const int rr = (lane >> 4) << 3;     // +8 for upper half-wave
  const int cc = lane & 15;
#pragma unroll
  for (int j = 0; j < 8; ++j) {
    const int col = j * 16 + cc;
    const float bv = bias ? bias[col] : 0.f;
#pragma unroll
    for (int r = 0; r < 8; ++r) {
      const size_t row = row0 + wv * 16 + r + rr;
      C[row * 128 + col] = acc[j][r] + bv;
    }
  }
}

// ---------------------------------------------------------------------------
// im2col for non-overlapping stride-S SxS patches of x (B,4096,128) -> bf16
// out[(b*m+p)*Kd + c*S*S + i*S + j] = x[b, (ph*S+i)*64 + (pw*S+j), c]
// ---------------------------------------------------------------------------
__global__ void im2col_patch(const float* __restrict__ x, __bf16* __restrict__ out,
                             int S, int PW, int m, size_t total) {
  size_t idx = (size_t)blockIdx.x * blockDim.x + threadIdx.x;
  if (idx >= total) return;
  const int Kd = 128 * S * S;
  const int kk = (int)(idx % Kd);
  const size_t rowp = idx / Kd;
  const int b = (int)(rowp / m), p = (int)(rowp % m);
  const int ph = p / PW, pw = p % PW;
  const int c = kk / (S * S);
  const int t = kk % (S * S);
  const int i = t / S, j = t % S;
  const size_t n = (size_t)(ph * S + i) * 64 + (pw * S + j);
  out[idx] = (__bf16)x[((size_t)b * 4096 + n) * 128 + c];
}

// q (B,4096,128) f32 -> (B,4,4096,32) bf16 with softmax scale folded in
__global__ void q_post(const float* __restrict__ qf, __bf16* __restrict__ qb,
                       float scale, size_t total) {
  size_t idx = (size_t)blockIdx.x * blockDim.x + threadIdx.x;
  if (idx >= total) return;
  const int hd = (int)(idx % 128);
  const size_t bn = idx / 128;
  const int n = (int)(bn % 4096);
  const int b = (int)(bn / 4096);
  const int h = hd >> 5, dd = hd & 31;
  qb[(((size_t)b * 4 + h) * 4096 + n) * 32 + dd] = (__bf16)(qf[idx] * scale);
}

// LayerNorm(+exact GELU) over C=128 per row; one block per row.
__global__ __launch_bounds__(128)
void ln_gelu(const float* __restrict__ t, const float* __restrict__ g,
             const float* __restrict__ be, __bf16* __restrict__ out) {
  __shared__ float red[128];
  const int row = blockIdx.x, tid = threadIdx.x;
  const float v = t[(size_t)row * 128 + tid];
  red[tid] = v; __syncthreads();
  for (int s = 64; s > 0; s >>= 1) { if (tid < s) red[tid] += red[tid + s]; __syncthreads(); }
  const float mean = red[0] * (1.f / 128.f);
  __syncthreads();
  const float d = v - mean;
  red[tid] = d * d; __syncthreads();
  for (int s = 64; s > 0; s >>= 1) { if (tid < s) red[tid] += red[tid + s]; __syncthreads(); }
  const float var = red[0] * (1.f / 128.f);
  const float y = d * rsqrtf(var + 1e-5f) * g[tid] + be[tid];
  const float ge = 0.5f * y * (1.f + erff(y * 0.70710678118654752f));
  out[(size_t)row * 128 + tid] = (__bf16)ge;
}

// kv (B,m,2,2,32) f32 -> k bf16 (B,2,m,32), v f32 (B,2,m,32)
__global__ void kv_split(const float* __restrict__ kv, __bf16* __restrict__ kb,
                         float* __restrict__ vf, int M, size_t total) {
  size_t idx = (size_t)blockIdx.x * blockDim.x + threadIdx.x;
  if (idx >= total) return;
  const int col = (int)(idx % 128);
  const size_t rp = idx / 128;
  const int b = (int)(rp / M), p = (int)(rp % M);
  const int two = col >> 6, h = (col >> 5) & 1, dd = col & 31;
  const size_t o = (((size_t)b * 2 + h) * M + p) * 32 + dd;
  const float v = kv[idx];
  if (two == 0) kb[o] = (__bf16)v; else vf[o] = v;
}

// depthwise 3x3 conv (pad 1) on v grid + residual add -> bf16
__global__ void local_conv(const float* __restrict__ vf, const float* __restrict__ w,
                           const float* __restrict__ bias, __bf16* __restrict__ vb,
                           int HP, int M, size_t total) {
  size_t idx = (size_t)blockIdx.x * blockDim.x + threadIdx.x;
  if (idx >= total) return;
  const int dd = (int)(idx & 31);
  const size_t r = idx >> 5;
  const int p = (int)(r % M);
  const size_t bh = r / M;
  const int h = (int)(bh % 2);
  const int y = p / HP, xx = p % HP;
  const int ch = h * 32 + dd;
  float s = bias[ch];
#pragma unroll
  for (int ky = 0; ky < 3; ++ky)
#pragma unroll
    for (int kx = 0; kx < 3; ++kx) {
      const int yy = y + ky - 1, xz = xx + kx - 1;
      if (yy >= 0 && yy < HP && xz >= 0 && xz < HP)
        s += w[ch * 9 + ky * 3 + kx] * vf[(bh * M + (size_t)yy * HP + xz) * 32 + dd];
    }
  vb[idx] = (__bf16)(vf[idx] + s);
}

// ---------------------------------------------------------------------------
// Fused attention: per (b, head, 64 query rows): S = q*k^T (WMMA),
// softmax in LDS (probs bf16 written in-place over f32 scores), out = P*V
// (WMMA), written into the concat buffer xcat (B*4096, 128) bf16.
// 4 waves/block, one 16-row q tile per wave. K/V for the head live in LDS.
// ---------------------------------------------------------------------------
template <int MKV>
__global__ __launch_bounds__(128)
void attn_fused(const __bf16* __restrict__ q, const __bf16* __restrict__ kb,
                const __bf16* __restrict__ vb, __bf16* __restrict__ xcat,
                int branch) {
  __shared__ __bf16 Ks[MKV * 32];
  __shared__ __bf16 Vs[MKV * 32];
  __shared__ __bf16 Qs[64 * 32];
  __shared__ float  Ss[4][16 * MKV];

  const int tid = threadIdx.x, lane = tid & 31, wv = tid >> 5;
  const int nb = blockIdx.x;          // 64-row query tile index
  const int h  = blockIdx.y;          // head within branch (0..1)
  const int b  = blockIdx.z;
  const int head = branch * 2 + h;

  const size_t kvbase = (((size_t)b * 2 + h) * MKV) * 32;
  for (int i = tid; i < MKV * 32; i += 128) { Ks[i] = kb[kvbase + i]; Vs[i] = vb[kvbase + i]; }
  const size_t qbase = (((size_t)b * 4 + head) * 4096 + (size_t)nb * 64) * 32;
  for (int i = tid; i < 64 * 32; i += 128) Qs[i] = q[qbase + i];
  __syncthreads();

  const v16bf aq = frag_A_rowmajor(&Qs[wv * 16 * 32], 32, lane);
  float* Sw = &Ss[wv][0];
  const int rr = (lane >> 4) << 3;
  const int cc = lane & 15;

  // S = q * k^T  (k rows loaded A-style give k^T as the B operand)
#pragma unroll
  for (int mt = 0; mt < MKV / 16; ++mt) {
    v16bf bk = frag_A_rowmajor(&Ks[mt * 16 * 32], 32, lane);
    v8f s = vzero8();
    s = wmma_bf16(aq, bk, s);
#pragma unroll
    for (int r = 0; r < 8; ++r) Sw[(size_t)(r + rr) * MKV + mt * 16 + cc] = s[r];
  }
  __syncthreads();

  // Row softmax: lanes 0..15 each own one row. bf16 probs overwrite the f32
  // row in place (safe: p[i] clobbers s[i>>1], already consumed).
  if (lane < 16) {
    float* srow = &Sw[(size_t)lane * MKV];
    __bf16* prow = (__bf16*)srow;
    float mx = -1e30f;
    for (int i = 0; i < MKV; ++i) mx = fmaxf(mx, srow[i]);
    float sum = 0.f;
    for (int i = 0; i < MKV; ++i) { float e = expf(srow[i] - mx); sum += e; srow[i] = e; }
    const float inv = 1.f / sum;
    for (int i = 0; i < MKV; ++i) prow[i] = (__bf16)(srow[i] * inv);
  }
  __syncthreads();

  // out(16x32) = P(16xMKV) * V(MKVx32); P rows have stride 2*MKV bf16 elems.
  v8f acc0 = vzero8(), acc1 = vzero8();
  const __bf16* Pbase = (const __bf16*)Sw;
#pragma unroll
  for (int mt = 0; mt < MKV / 32; ++mt) {
    v16bf ap  = frag_A_rowmajor(Pbase + mt * 32, 2 * MKV, lane);
    v16bf bv0 = frag_B_kxn(&Vs[(mt * 32) * 32 + 0],  32, lane);
    v16bf bv1 = frag_B_kxn(&Vs[(mt * 32) * 32 + 16], 32, lane);
    acc0 = wmma_bf16(ap, bv0, acc0);
    acc1 = wmma_bf16(ap, bv1, acc1);
  }

  const int colbase = branch * 64 + h * 32;
  const size_t rowg = (size_t)b * 4096 + (size_t)nb * 64 + wv * 16;
#pragma unroll
  for (int r = 0; r < 8; ++r) {
    const size_t rw = (rowg + r + rr) * 128;
    xcat[rw + colbase + cc]      = (__bf16)acc0[r];
    xcat[rw + colbase + 16 + cc] = (__bf16)acc1[r];
  }
}

// ---------------------------------------------------------------------------
// Host-side launch
// ---------------------------------------------------------------------------
extern "C" void kernel_launch(void* const* d_in, const int* in_sizes, int n_in,
                              void* d_out, int out_size, void* d_ws, size_t ws_size,
                              hipStream_t stream) {
  const float* x      = (const float*)d_in[0];
  const float* q_w    = (const float*)d_in[1];
  const float* sr1_w  = (const float*)d_in[2];
  const float* sr1_b  = (const float*)d_in[3];
  const float* n1_g   = (const float*)d_in[4];
  const float* n1_b   = (const float*)d_in[5];
  const float* sr2_w  = (const float*)d_in[6];
  const float* sr2_b  = (const float*)d_in[7];
  const float* n2_g   = (const float*)d_in[8];
  const float* n2_b   = (const float*)d_in[9];
  const float* kv1_w  = (const float*)d_in[10];
  const float* kv2_w  = (const float*)d_in[11];
  const float* lc1_w  = (const float*)d_in[12];
  const float* lc1_b  = (const float*)d_in[13];
  const float* lc2_w  = (const float*)d_in[14];
  const float* lc2_b  = (const float*)d_in[15];
  const float* proj_w = (const float*)d_in[16];
  const float* proj_b = (const float*)d_in[17];
  float* out = (float*)d_out;

  const int Bn = 16, Nn = 4096;
  const size_t BN = (size_t)Bn * Nn;           // 65536

  size_t off = 0;
  char* wsb = (char*)d_ws;
  auto take = [&](size_t bytes) -> char* {
    char* p = wsb + off;
    off += (bytes + 255) & ~(size_t)255;
    return p;
  };

  float*  GBUF = (float*)take(BN * 128 * 4);                 // qf32, later im2col A1+A2
  __bf16* qb   = (__bf16*)take((size_t)Bn * 4 * Nn * 32 * 2);
  float*  t1f  = (float*) take((size_t)Bn * 64 * 128 * 4);
  __bf16* t1b  = (__bf16*)take((size_t)Bn * 64 * 128 * 2);
  float*  kv1f = (float*) take((size_t)Bn * 64 * 128 * 4);
  __bf16* k1b  = (__bf16*)take((size_t)Bn * 2 * 64 * 32 * 2);
  float*  v1f  = (float*) take((size_t)Bn * 2 * 64 * 32 * 4);
  __bf16* v1b  = (__bf16*)take((size_t)Bn * 2 * 64 * 32 * 2);
  float*  t2f  = (float*) take((size_t)Bn * 256 * 128 * 4);
  __bf16* t2b  = (__bf16*)take((size_t)Bn * 256 * 128 * 2);
  float*  kv2f = (float*) take((size_t)Bn * 256 * 128 * 4);
  __bf16* k2b  = (__bf16*)take((size_t)Bn * 2 * 256 * 32 * 2);
  float*  v2f  = (float*) take((size_t)Bn * 2 * 256 * 32 * 4);
  __bf16* v2b  = (__bf16*)take((size_t)Bn * 2 * 256 * 32 * 2);
  __bf16* xcat = (__bf16*)take(BN * 128 * 2);

  float*  qf32 = GBUF;
  __bf16* A1   = (__bf16*)GBUF;                       // 1024 x 8192 bf16 (16.8MB)
  __bf16* A2   = A1 + (size_t)1024 * 8192;            // 4096 x 2048 bf16 (16.8MB)

  const float scale = 0.17677669529663687f;           // 32^-0.5

  // 1) q = x @ q_w^T
  gemm_n128<float, float><<<(int)(BN / 64), 128, 0, stream>>>(x, q_w, nullptr, qf32, 128);
  {
    size_t total = BN * 128;
    q_post<<<(int)((total + 255) / 256), 256, 0, stream>>>(qf32, qb, scale, total);
  }

  // 2) patch embeds (non-overlapping conv == GEMM over im2col)
  {
    size_t total1 = (size_t)Bn * 64 * 8192;
    im2col_patch<<<(int)((total1 + 255) / 256), 256, 0, stream>>>(x, A1, 8, 8, 64, total1);
    size_t total2 = (size_t)Bn * 256 * 2048;
    im2col_patch<<<(int)((total2 + 255) / 256), 256, 0, stream>>>(x, A2, 4, 16, 256, total2);
  }
  gemm_n128<__bf16, float><<<(Bn * 64) / 64, 128, 0, stream>>>(A1, sr1_w, sr1_b, t1f, 8192);
  gemm_n128<__bf16, float><<<(Bn * 256) / 64, 128, 0, stream>>>(A2, sr2_w, sr2_b, t2f, 2048);

  // 3) LayerNorm + GELU
  ln_gelu<<<Bn * 64, 128, 0, stream>>>(t1f, n1_g, n1_b, t1b);
  ln_gelu<<<Bn * 256, 128, 0, stream>>>(t2f, n2_g, n2_b, t2b);

  // 4) kv projections
  gemm_n128<__bf16, float><<<(Bn * 64) / 64, 128, 0, stream>>>(t1b, kv1_w, nullptr, kv1f, 128);
  gemm_n128<__bf16, float><<<(Bn * 256) / 64, 128, 0, stream>>>(t2b, kv2_w, nullptr, kv2f, 128);
  {
    size_t tot1 = (size_t)Bn * 64 * 128;
    kv_split<<<(int)((tot1 + 255) / 256), 256, 0, stream>>>(kv1f, k1b, v1f, 64, tot1);
    size_t tot2 = (size_t)Bn * 256 * 128;
    kv_split<<<(int)((tot2 + 255) / 256), 256, 0, stream>>>(kv2f, k2b, v2f, 256, tot2);
  }

  // 5) depthwise local conv on v + residual
  {
    size_t tot1 = (size_t)Bn * 2 * 64 * 32;
    local_conv<<<(int)((tot1 + 255) / 256), 256, 0, stream>>>(v1f, lc1_w, lc1_b, v1b, 8, 64, tot1);
    size_t tot2 = (size_t)Bn * 2 * 256 * 32;
    local_conv<<<(int)((tot2 + 255) / 256), 256, 0, stream>>>(v2f, lc2_w, lc2_b, v2b, 16, 256, tot2);
  }

  // 6) fused attention per branch -> concat buffer
  attn_fused<64> <<<dim3(Nn / 64, 2, Bn), 128, 0, stream>>>(qb, k1b, v1b, xcat, 0);
  attn_fused<256><<<dim3(Nn / 64, 2, Bn), 128, 0, stream>>>(qb, k2b, v2b, xcat, 1);

  // 7) final projection -> fp32 output
  gemm_n128<__bf16, float><<<(int)(BN / 64), 128, 0, stream>>>(xcat, proj_w, proj_b, out, 128);
}
